// SpatialTemporalLayer_28329604284817
// MI455X (gfx1250) — matrix operand order repaired and checked by reading, not compiled, not run on hardware
//
#include <hip/hip_runtime.h>

typedef __attribute__((ext_vector_type(16))) __bf16 v16bf;
typedef __attribute__((ext_vector_type(8)))  float  v8f;
typedef unsigned int u32;
typedef __attribute__((ext_vector_type(4))) u32 u32x4;
typedef __attribute__((ext_vector_type(8))) u32 u32x8;

// ---------------------------------------------------------------- helpers ---

static __device__ __forceinline__ v8f wmma_bf16(v16bf a, v16bf b, v8f c) {
  // D = A(16x32 bf16) x B(32x16 bf16) + C(16x16 f32)
  return __builtin_amdgcn_wmma_f32_16x16x32_bf16(false, a, false, b, (short)0, c,
                                                 false, false);
}

static __device__ __forceinline__ void lds_fence() {
  asm volatile("s_wait_dscnt 0" ::: "memory");
}

// Branch-free fast transcendentals (v_exp_f32 / v_rcp_f32 / v_rsq_f32).
static __device__ __forceinline__ float fast_sigmoid(float x) {
  return __builtin_amdgcn_rcpf(1.0f + __expf(-x));
}
static __device__ __forceinline__ float fast_tanh(float x) {
  // 1 - 2/(e^{2x}+1); exp overflow/underflow saturates to +-1 correctly.
  return 1.0f - 2.0f * __builtin_amdgcn_rcpf(__expf(2.0f * x) + 1.0f);
}

static __device__ __forceinline__ v16bf make_frag(u32x4 lo, u32x4 hi) {
  u32x8 r = __builtin_shufflevector(lo, hi, 0, 1, 2, 3, 4, 5, 6, 7);
  return __builtin_bit_cast(v16bf, r);
}

// A operand (16x32, bf16) from row-major [m][k] buffer (k contiguous).
// lane L: row m = m0 + (L&15), k-base = (L>>4)*8 ; VGPR v: k = (v>>2)*16 + kb + (v&3)*2 (+1)
static __device__ __forceinline__ v16bf load_a_rowmajor(const __bf16* p, int pitch,
                                                        int m0, int k0) {
  int lane = threadIdx.x & 31;
  const __bf16* rowp = p + (size_t)(m0 + (lane & 15)) * pitch + k0 + ((lane >> 4) << 3);
  u32x4 lo = *(const u32x4*)rowp;        // k = kb .. kb+7
  u32x4 hi = *(const u32x4*)(rowp + 16); // k = 16+kb .. 16+kb+7
  return make_frag(lo, hi);
}

// B operand (32x16, bf16) from [n][k] buffer (k contiguous per column n).
// lane L: n = n0 + (L&15), k-base = (L>>4)*8 ; VGPR v: k = (v>>2)*16 + kb + (v&3)*2 (+1)
static __device__ __forceinline__ v16bf load_b_nk(const __bf16* p, int pitch,
                                                  int n0, int k0) {
  int lane = threadIdx.x & 31;
  const __bf16* q = p + (size_t)(n0 + (lane & 15)) * pitch + k0 + ((lane >> 4) << 3);
  u32x4 lo = *(const u32x4*)q;
  u32x4 hi = *(const u32x4*)(q + 16);
  return make_frag(lo, hi);
}

// A operand from k-major [k][col] buffer (strided u16 gathers).
static __device__ __forceinline__ v16bf load_a_kmajor(const __bf16* p, int pitch,
                                                      int m0, int k0) {
  int lane = threadIdx.x & 31;
  int col = m0 + (lane & 15);
  int kb = (lane >> 4) << 3;
  u32 r[8];
#pragma unroll
  for (int v = 0; v < 8; ++v) {
    int k = k0 + ((v >> 2) << 4) + kb + ((v & 3) << 1);
    u32 lo = (u32)__builtin_bit_cast(unsigned short, p[(size_t)k * pitch + col]);
    u32 hi = (u32)__builtin_bit_cast(unsigned short, p[(size_t)(k + 1) * pitch + col]);
    r[v] = lo | (hi << 16);
  }
  u32x8 rv = {r[0], r[1], r[2], r[3], r[4], r[5], r[6], r[7]};
  return __builtin_bit_cast(v16bf, rv);
}

static __device__ __forceinline__ u32 pack_bf16x2(float a, float b) {
  u32 lo = (u32)__builtin_bit_cast(unsigned short, (__bf16)a);
  u32 hi = (u32)__builtin_bit_cast(unsigned short, (__bf16)b);
  return lo | (hi << 16);
}

// A operand from per-wave f32 scratch [16][pitch] (row-major in k), f32->bf16.
static __device__ __forceinline__ v16bf load_a_scratch(const float* s, int pitch,
                                                       int k0) {
  int lane = threadIdx.x & 31;
  int row = lane & 15;
  int kb = (lane >> 4) << 3;
  u32 r[8];
#pragma unroll
  for (int v = 0; v < 8; ++v) {
    int k = k0 + ((v >> 2) << 4) + kb + ((v & 3) << 1);
    r[v] = pack_bf16x2(s[row * pitch + k], s[row * pitch + k + 1]);
  }
  u32x8 rv = {r[0], r[1], r[2], r[3], r[4], r[5], r[6], r[7]};
  return __builtin_bit_cast(v16bf, rv);
}

// ------------------------------------------------------------ prep kernels ---

// Convert the six needed N x N diffusion matrices to bf16:
// p0=Pf^1 p1=Pf^2 p2=Pb^1 p3=Pb^2 p4=A^1 p5=A^2   (each 512*512)
__global__ __launch_bounds__(256) void stl_prep_p(const float* __restrict__ Pf,
                                                  const float* __restrict__ Pb,
                                                  const float* __restrict__ Aa,
                                                  __bf16* __restrict__ dst) {
  int idx = blockIdx.x * 256 + threadIdx.x;
  if (idx >= 6 * 512 * 512) return;
  int p = idx >> 18;
  int rem = idx & 262143;
  float v;
  if (p < 2)      v = Pf[(size_t)(1 + p) * 262144 + rem];
  else if (p < 4) v = Pb[(size_t)(p - 1) * 262144 + rem];
  else            v = Aa[(size_t)(p - 3) * 262144 + rem];
  dst[idx] = (__bf16)v;
}

// Effective GCN weights (transposed, [co][ci]) and TCN taps ([co][ci]).
__global__ __launch_bounds__(256) void stl_prep_w(const float* __restrict__ Wf,
                                                  const float* __restrict__ Wb,
                                                  const float* __restrict__ fw,
                                                  const float* __restrict__ gw,
                                                  const float* __restrict__ sw,
                                                  const float* __restrict__ alpha,
                                                  __bf16* __restrict__ wt,
                                                  __bf16* __restrict__ tcw) {
  int idx = blockIdx.x * 256 + threadIdx.x;
  if (idx >= 12 * 4096) return;
  float a = fast_sigmoid(alpha[0]);
  int q = idx >> 12, r = idx & 4095;
  int co = r >> 6, ci = r & 63;
  int tp = ci * 64 + co;  // source index for transpose ( weights are [k][ci][co] )
  float v;
  switch (q) {
    case 0:  v = Wf[tp] + Wb[tp]; break;                                   // W0
    case 1:  v = a * Wf[4096 + tp]; break;                                 // Pf^1
    case 2:  v = a * Wf[8192 + tp]; break;                                 // Pf^2
    case 3:  v = a * Wb[4096 + tp]; break;                                 // Pb^1
    case 4:  v = a * Wb[8192 + tp]; break;                                 // Pb^2
    case 5:  v = (1.f - a) * (Wf[4096 + tp] + Wb[4096 + tp]); break;       // A^1
    case 6:  v = (1.f - a) * (Wf[8192 + tp] + Wb[8192 + tp]); break;       // A^2
    case 7:  v = fw[r * 2]; break;      // filter tap0 (x[t-1])
    case 8:  v = fw[r * 2 + 1]; break;  // filter tap1 (x[t])
    case 9:  v = gw[r * 2]; break;      // gate tap0
    case 10: v = gw[r * 2 + 1]; break;  // gate tap1
    default: v = sw[r]; break;          // skip 1x1
  }
  if (q < 7) wt[q * 4096 + r] = (__bf16)v;
  else       tcw[(q - 7) * 4096 + r] = (__bf16)v;
}

// ------------------------------------------------------------- TCN kernel ---
// One workgroup per (b, n): 64(ch) x 64(t) tile. Computes
//   out = tanh(conv_f) * sigmoid(conv_g), skip = 1x1(out), x_temporal = x + out
// x_temporal written in bf16 (bt, c, j=n) layout for the GCN kernel.
__global__ __launch_bounds__(256) void stl_tcn(const float* __restrict__ x,
                                               const __bf16* __restrict__ tw,
                                               const float* __restrict__ fb,
                                               const float* __restrict__ gb,
                                               const float* __restrict__ sb,
                                               __bf16* __restrict__ xt_out,
                                               float* __restrict__ skip_out) {
  __shared__ __bf16 Xl[65][72];    // [t+1][c], row 0 = causal zero pad (pitch 72 => no bank conflicts)
  __shared__ __bf16 OutL[64][72];  // [t][c]
  int bx = blockIdx.x;
  int b = bx >> 9, n = bx & 511;
  int tid = threadIdx.x;
  int w = tid >> 5, lane = tid & 31;
  int nn = lane & 15, hib = (lane >> 4) << 3;

  for (int i = tid; i < 4096; i += 256) {
    int c = i >> 6, t = i & 63;
    float v = x[(((size_t)b * 64 + c) * 512 + n) * 64 + t];
    Xl[t + 1][c] = (__bf16)v;
  }
  if (tid < 64) Xl[0][tid] = (__bf16)0.0f;
  __syncthreads();

  const __bf16* wf0 = tw;
  const __bf16* wf1 = tw + 4096;
  const __bf16* wg0 = tw + 8192;
  const __bf16* wg1 = tw + 12288;
  const __bf16* wsk = tw + 16384;

  // ---- phase 1: gated conv + x_temporal -----------------------------------
#pragma unroll
  for (int i = 0; i < 2; ++i) {
    int tau = w * 2 + i;
    int co0 = (tau >> 2) << 4, t0 = (tau & 3) << 4;
    v8f aF = {0, 0, 0, 0, 0, 0, 0, 0};
    v8f aG = {0, 0, 0, 0, 0, 0, 0, 0};
#pragma unroll
    for (int kt = 0; kt < 64; kt += 32) {
      v16bf b0 = load_b_nk(&Xl[0][0], 72, t0, kt);  // rows t0+nn   -> x[t-1]
      v16bf b1 = load_b_nk(&Xl[1][0], 72, t0, kt);  // rows t0+nn+1 -> x[t]
      aF = wmma_bf16(load_a_rowmajor(wf0, 64, co0, kt), b0, aF);
      aF = wmma_bf16(load_a_rowmajor(wf1, 64, co0, kt), b1, aF);
      aG = wmma_bf16(load_a_rowmajor(wg0, 64, co0, kt), b0, aG);
      aG = wmma_bf16(load_a_rowmajor(wg1, 64, co0, kt), b1, aG);
    }
    int tloc = t0 + nn;
    size_t bt = (size_t)b * 64 + tloc;
#pragma unroll
    for (int r = 0; r < 8; ++r) {
      int co = co0 + r + hib;
      float f = fast_tanh(aF[r] + fb[co]);
      float g = fast_sigmoid(aG[r] + gb[co]);
      float o = f * g;
      float xv = (float)Xl[tloc + 1][co];
      xt_out[(bt * 64 + co) * 512 + n] = (__bf16)(xv + o);
      OutL[tloc][co] = (__bf16)o;
    }
  }
  __syncthreads();

  // ---- phase 2: skip = 1x1 conv of out ------------------------------------
#pragma unroll
  for (int i = 0; i < 2; ++i) {
    int tau = w * 2 + i;
    int co0 = (tau >> 2) << 4, t0 = (tau & 3) << 4;
    v8f aS = {0, 0, 0, 0, 0, 0, 0, 0};
#pragma unroll
    for (int kt = 0; kt < 64; kt += 32)
      aS = wmma_bf16(load_a_rowmajor(wsk, 64, co0, kt),
                     load_b_nk(&OutL[0][0], 72, t0, kt), aS);
#pragma unroll
    for (int r = 0; r < 8; ++r) {
      int co = co0 + r + hib;
      skip_out[(((size_t)b * 64 + co) * 512 + n) * 64 + t0 + nn] = aS[r] + sb[co];
    }
  }
}

// ------------------------------------------------------------- GCN kernel ---
// One workgroup per bt. x_gcn = X*W0 + sum_p (Pp[rows] @ X) * Wp' + bias, then
// LayerNorm over C and transpose-store to (B,C,N,T).
#define XT_PITCH 520  // bf16 elements; 1040B rows => conflict-free 16B B reads
#define SCR_PITCH 66

__global__ __launch_bounds__(256) void stl_gcn(const __bf16* __restrict__ xt_all,
                                               const __bf16* __restrict__ Pm,
                                               const __bf16* __restrict__ Wt,
                                               const float* __restrict__ gbias,
                                               const float* __restrict__ gamma,
                                               const float* __restrict__ beta,
                                               float* __restrict__ out_x) {
  extern __shared__ char smem[];
  __bf16* Xt = (__bf16*)smem;                           // [64][XT_PITCH]
  float* scrbase = (float*)(smem + 64 * XT_PITCH * 2);  // 8 x [16][SCR_PITCH]
  int bt = blockIdx.x;
  int b = bt >> 6, t = bt & 63;
  int tid = threadIdx.x, w = tid >> 5, lane = tid & 31;
  int nn = lane & 15, hib = (lane >> 4) << 3;
  float* scr = scrbase + w * (16 * SCR_PITCH);

  {  // load X_bt (512 x 64) as Xt[c][j], bf16, coalesced u32 copies
    const u32* src = (const u32*)(xt_all + (size_t)bt * 32768);
    for (int i = tid; i < 16384; i += 256) {
      int c = i >> 8, jp = i & 255;
      *(u32*)(Xt + c * XT_PITCH + jp * 2) = src[i];
    }
  }
  __syncthreads();

  int m0w = w << 6;  // each wave owns 64 output rows
#pragma unroll 1
  for (int half = 0; half < 2; ++half) {
    int mbase = m0w + (half << 5);
    v8f acc[2][4];
    // ---- k=0 term: X @ W0 (A gathered k-major from Xt) ----
#pragma unroll
    for (int s = 0; s < 2; ++s) {
      v16bf a0 = load_a_kmajor(Xt, XT_PITCH, mbase + s * 16, 0);
      v16bf a1 = load_a_kmajor(Xt, XT_PITCH, mbase + s * 16, 32);
#pragma unroll
      for (int ct = 0; ct < 4; ++ct) {
        v8f c = {0, 0, 0, 0, 0, 0, 0, 0};
        c = wmma_bf16(a0, load_b_nk(Wt, 64, ct * 16, 0), c);
        c = wmma_bf16(a1, load_b_nk(Wt, 64, ct * 16, 32), c);
        acc[s][ct] = c;
      }
    }
    // ---- six diffusion terms ----
#pragma unroll 1
    for (int p = 0; p < 6; ++p) {
      const __bf16* Pp = Pm + (size_t)p * 262144;
      v8f accp[2][4];
#pragma unroll
      for (int s = 0; s < 2; ++s)
#pragma unroll
        for (int ct = 0; ct < 4; ++ct) accp[s][ct] = (v8f){0, 0, 0, 0, 0, 0, 0, 0};
#pragma unroll 2
      for (int jt = 0; jt < 16; ++jt) {
        int j0 = jt << 5;
        v16bf bx0 = load_b_nk(Xt, XT_PITCH, 0, j0);
        v16bf bx1 = load_b_nk(Xt, XT_PITCH, 16, j0);
        v16bf bx2 = load_b_nk(Xt, XT_PITCH, 32, j0);
        v16bf bx3 = load_b_nk(Xt, XT_PITCH, 48, j0);
        v16bf a0 = load_a_rowmajor(Pp, 512, mbase, j0);
        v16bf a1 = load_a_rowmajor(Pp, 512, mbase + 16, j0);
        accp[0][0] = wmma_bf16(a0, bx0, accp[0][0]);
        accp[0][1] = wmma_bf16(a0, bx1, accp[0][1]);
        accp[0][2] = wmma_bf16(a0, bx2, accp[0][2]);
        accp[0][3] = wmma_bf16(a0, bx3, accp[0][3]);
        accp[1][0] = wmma_bf16(a1, bx0, accp[1][0]);
        accp[1][1] = wmma_bf16(a1, bx1, accp[1][1]);
        accp[1][2] = wmma_bf16(a1, bx2, accp[1][2]);
        accp[1][3] = wmma_bf16(a1, bx3, accp[1][3]);
      }
      // apply Wp': C-layout -> bf16 A-layout via per-wave LDS scratch
      const __bf16* Wp = Wt + (size_t)(1 + p) * 4096;
#pragma unroll 1
      for (int s = 0; s < 2; ++s) {
        lds_fence();
#pragma unroll
        for (int ct = 0; ct < 4; ++ct)
#pragma unroll
          for (int r = 0; r < 8; ++r)
            scr[(r + hib) * SCR_PITCH + ct * 16 + nn] = accp[s][ct][r];
        lds_fence();
        v16bf aw0 = load_a_scratch(scr, SCR_PITCH, 0);
        v16bf aw1 = load_a_scratch(scr, SCR_PITCH, 32);
#pragma unroll
        for (int ct = 0; ct < 4; ++ct) {
          acc[s][ct] = wmma_bf16(aw0, load_b_nk(Wp, 64, ct * 16, 0), acc[s][ct]);
          acc[s][ct] = wmma_bf16(aw1, load_b_nk(Wp, 64, ct * 16, 32), acc[s][ct]);
        }
      }
    }
    // ---- bias + LayerNorm + transpose store ----
#pragma unroll 1
    for (int s = 0; s < 2; ++s) {
      lds_fence();
#pragma unroll
      for (int ct = 0; ct < 4; ++ct)
#pragma unroll
        for (int r = 0; r < 8; ++r)
          scr[(r + hib) * SCR_PITCH + ct * 16 + nn] = acc[s][ct][r];
      lds_fence();
      int rr = lane & 15;
      int ch = (lane >> 4) << 5;  // lane halves split the 64 channels on output
      float sum = 0.f, sq = 0.f;
#pragma unroll 8
      for (int c = 0; c < 64; ++c) {
        float v = scr[rr * SCR_PITCH + c] + gbias[c];
        sum += v;
        sq += v * v;
      }
      float mu = sum * (1.0f / 64.0f);
      float var = sq * (1.0f / 64.0f) - mu * mu;
      float rs = __builtin_amdgcn_rsqf(var + 1e-5f);
      int nrow = mbase + s * 16 + rr;
#pragma unroll 8
      for (int c2 = 0; c2 < 32; ++c2) {
        int c = ch + c2;
        float v = scr[rr * SCR_PITCH + c] + gbias[c];
        out_x[(((size_t)b * 64 + c) * 512 + nrow) * 64 + t] =
            (v - mu) * rs * gamma[c] + beta[c];
      }
    }
  }
}

// ---------------------------------------------------------------- launcher ---

extern "C" void kernel_launch(void* const* d_in, const int* in_sizes, int n_in,
                              void* d_out, int out_size, void* d_ws, size_t ws_size,
                              hipStream_t stream) {
  const float* x      = (const float*)d_in[0];
  const float* Pf     = (const float*)d_in[1];
  const float* Pb     = (const float*)d_in[2];
  const float* Aa     = (const float*)d_in[3];
  const float* fw     = (const float*)d_in[4];
  const float* fb     = (const float*)d_in[5];
  const float* gw     = (const float*)d_in[6];
  const float* gb     = (const float*)d_in[7];
  const float* sw     = (const float*)d_in[8];
  const float* sb     = (const float*)d_in[9];
  const float* Wf     = (const float*)d_in[10];
  const float* Wb     = (const float*)d_in[11];
  const float* gbias  = (const float*)d_in[12];
  const float* gamma  = (const float*)d_in[13];
  const float* beta   = (const float*)d_in[14];
  const float* alpha  = (const float*)d_in[15];

  char* ws = (char*)d_ws;
  __bf16* ws_xt   = (__bf16*)ws;                  // 1024*64*512 bf16 = 64 MiB
  __bf16* ws_P    = (__bf16*)(ws + 67108864);     // 6*512*512 bf16  = 3 MiB
  __bf16* ws_wt   = (__bf16*)(ws + 70254592);     // 7*4096 bf16
  __bf16* ws_tcnw = (__bf16*)(ws + 70311936);     // 5*4096 bf16

  float* out_x    = (float*)d_out;                // (B,C,N,T) = 33554432 f32
  float* out_skip = out_x + 33554432;             // (B,C,N,T)

  stl_prep_p<<<6144, 256, 0, stream>>>(Pf, Pb, Aa, ws_P);
  stl_prep_w<<<192, 256, 0, stream>>>(Wf, Wb, fw, gw, sw, alpha, ws_wt, ws_tcnw);
  stl_tcn<<<8192, 256, 0, stream>>>(x, ws_tcnw, fb, gb, sb, ws_xt, out_skip);

  size_t shbytes = (size_t)64 * XT_PITCH * 2 + (size_t)8 * 16 * SCR_PITCH * 4; // ~98 KiB
  (void)hipFuncSetAttribute((const void*)stl_gcn,
                            hipFuncAttributeMaxDynamicSharedMemorySize, (int)shbytes);
  stl_gcn<<<1024, 256, shbytes, stream>>>(ws_xt, ws_P, ws_wt, gbias, gamma, beta, out_x);
}
